// DTNN_44195213476531
// MI455X (gfx1250) — compile-verified, matrix operand order checked by reading, not run on hardware
//
#include <hip/hip_runtime.h>
#include <hip/hip_bf16.h>
#include <math.h>

#define N_NODES   20000
#define N_EDGES   640000
#define BASIS     128
#define NUM_GAUSS 64
#define HIDDEN    256
#define NUM_GRAPHS 256
#define T_ITERS   3
#define EK_WAVES  4       // waves per block in edge kernel (128 threads)

typedef __attribute__((ext_vector_type(16))) __bf16 v16bf;
typedef __attribute__((ext_vector_type(8)))  __bf16 v8bf;
typedef __attribute__((ext_vector_type(8)))  float  v8f;
typedef __attribute__((ext_vector_type(4)))  unsigned int v4u;
typedef __attribute__((ext_vector_type(8)))  unsigned int v8u;

// ---------------- WMMA helpers (CDNA5 16x16x32 bf16, f32 accum) ----------------

__device__ __forceinline__ v8f wmma_bf16(v16bf a, v16bf b, v8f c) {
  // 8 args: (neg_a, A, neg_b, B, c_mod, C, reuse_a, reuse_b)
  return __builtin_amdgcn_wmma_f32_16x16x32_bf16(false, a, false, b, (short)0, c,
                                                 false, false);
}

__device__ __forceinline__ v16bf make16(v8bf lo, v8bf hi) {
  return __builtin_shufflevector(lo, hi, 0,1,2,3,4,5,6,7,8,9,10,11,12,13,14,15);
}

// A fragment (16x32 bf16 layout): lane = M (dup in hi half), element e<8 -> K=t*32+hi*8+e,
// e>=8 -> K=t*32+16+hi*8+(e-8).  Source row is fp32 row-major; convert to bf16.
__device__ __forceinline__ v16bf a_frag_f32(const float* row, int t, int hi) {
  const float* p0 = row + t * 32 + hi * 8;
  const float* p1 = p0 + 16;
  v16bf r;
#pragma unroll
  for (int e = 0; e < 8; ++e) {
    r[e]     = (__bf16)p0[e];
    r[e + 8] = (__bf16)p1[e];
  }
  return r;
}

// Same fragment from a bf16 row-major row (two 16B vector loads).
__device__ __forceinline__ v16bf a_frag_bf16(const __bf16* row, int t, int hi) {
  v8bf lo = *(const v8bf*)(row + t * 32 + hi * 8);
  v8bf hh = *(const v8bf*)(row + t * 32 + 16 + hi * 8);
  return make16(lo, hh);
}

// B fragment from a pre-tiled buffer (global or LDS): 16 contiguous bf16 per lane.
__device__ __forceinline__ v16bf b_frag(const __bf16* p) {
  v8bf lo = *(const v8bf*)p;
  v8bf hh = *(const v8bf*)(p + 8);
  return make16(lo, hh);
}

// ---------------- Weight pre-tiling ----------------
// For W [Nout, K] (we need B = W.T, i.e. B[k][n] = W[n][k]) build
// Btile[((j*tT + t)*32 + lane)*16 + e] = bf16(W[n=j*16+(lane&15), k(t,lane,e)])
__global__ void prep_b_kernel(const float* __restrict__ W, unsigned short* __restrict__ Bt,
                              int Kdim, int jTiles, int tTiles) {
  int idx = blockIdx.x * blockDim.x + threadIdx.x;
  int total = jTiles * tTiles * 512;
  if (idx >= total) return;
  int e    = idx & 15;
  int l    = (idx >> 4) & 31;
  int tile = idx >> 9;
  int t = tile % tTiles;
  int j = tile / tTiles;
  int hi = l >> 4;
  int i  = e >> 1;
  int h  = e & 1;
  int k  = t * 32 + ((i >= 4) ? 16 : 0) + hi * 8 + (i & 3) * 2 + h;
  int n  = j * 16 + (l & 15);
  __bf16 v = (__bf16)W[(size_t)n * Kdim + k];
  Bt[idx] = *(unsigned short*)&v;
}

// ---------------- TDM probe: NULL-descriptor tensor_load_to_lds ----------------
// D# group0 count bits[1:0]==0 => NULL tensor (architectural no-op), but it
// exercises the Tensor Data Mover issue path + TENSORcnt.
__global__ void tdm_probe_kernel() {
  v4u g0 = {0u, 0u, 0u, 2u << 30};  // type=2 ("image"), count=0 -> NULL tensor
  v8u g1 = {0u, 0u, 0u, 0u, 0u, 0u, 0u, 0u};
  asm volatile("tensor_load_to_lds %0, %1" :: "s"(g0), "s"(g1) : "memory");
  __builtin_amdgcn_s_wait_tensorcnt(0);
}

// ---------------- C = embed[Z] ----------------
__global__ void gather_embed_kernel(const int* __restrict__ Z, const float* __restrict__ embed,
                                    float* __restrict__ C) {
  int i = blockIdx.x * blockDim.x + threadIdx.x;
  if (i >= N_NODES * BASIS) return;
  int node = i >> 7;
  int c    = i & 127;
  C[i] = embed[(size_t)Z[node] * BASIS + c];
}

// ---------------- CFb = bf16(C @ cfW.T + cfb), node-level GEMM ----------------
__global__ void cf_kernel(const float* __restrict__ C, const float* __restrict__ cfb,
                          const unsigned short* __restrict__ Bcf_u,
                          unsigned short* __restrict__ CFb_u) {
  const __bf16* Bcf = (const __bf16*)Bcf_u;
  __bf16* CFb = (__bf16*)CFb_u;
  int wave = (blockIdx.x * blockDim.x + threadIdx.x) >> 5;
  int lane = threadIdx.x & 31;
  int base = wave * 16;
  if (base >= N_NODES) return;            // wave-uniform guard (EXEC stays all-ones)
  int hi = lane >> 4, lo = lane & 15;

  const float* row = C + (size_t)(base + lo) * BASIS;
  v16bf a[4];
#pragma unroll
  for (int t = 0; t < 4; ++t) a[t] = a_frag_f32(row, t, hi);

#pragma unroll
  for (int j = 0; j < 8; ++j) {
    v8f acc = {0.f, 0.f, 0.f, 0.f, 0.f, 0.f, 0.f, 0.f};
#pragma unroll
    for (int t = 0; t < 4; ++t)
      acc = wmma_bf16(a[t], b_frag(Bcf + ((size_t)(j * 4 + t) * 32 + lane) * 16), acc);
    float bias = cfb[j * 16 + lo];
#pragma unroll
    for (int r = 0; r < 8; ++r) {
      int m = r + hi * 8;                 // D layout: lane holds (M=r+8*hi, N=lo)
      CFb[(size_t)(base + m) * BASIS + j * 16 + lo] = (__bf16)(acc[r] + bias);
    }
  }
}

// ---------------- Fused edge kernel (M-block = 2: 32 edges per wave) ----------------
// d_feat = edge_attr@dfW.T+dfb (WMMA, B from LDS via async copy) -> LDS transpose ->
// p = CF[src] * d_feat -> m = tanh(p @ fcW.T) (WMMA) -> atomicAdd into C[dst].
__global__ void edge_kernel(const int* __restrict__ src, const int* __restrict__ dst,
                            const float* __restrict__ edge_attr, const float* __restrict__ dfb,
                            const unsigned short* __restrict__ Bdf_u,
                            const unsigned short* __restrict__ Bfc_u,
                            const unsigned short* __restrict__ CFb_u,
                            float* __restrict__ C) {
  __shared__ __align__(16) unsigned short s_df[EK_WAVES * 32 * BASIS];  // 4 waves * 8KB
  __shared__ __align__(16) unsigned short s_bdf[8 * 2 * 512];           // 16KB dfW tiles
  const __bf16* Bfc = (const __bf16*)Bfc_u;
  const __bf16* CFb = (const __bf16*)CFb_u;

  int tid = threadIdx.x;

  // --- Async copy of the dfW B-tiles into LDS (GLOBAL_LOAD_ASYNC_TO_LDS_B128,
  //     tracked by ASYNCcnt). 16KB = 1024 x 16B chunks over 128 threads.
  {
    unsigned lbase = (unsigned)(uintptr_t)(void*)&s_bdf[0];  // flat->LDS offset (addr[31:0])
    const char* g = (const char*)Bdf_u;
#pragma unroll
    for (int c = 0; c < 8; ++c) {
      int chunk = tid + c * 128;
      unsigned laddr = lbase + chunk * 16;
      unsigned long long gaddr = (unsigned long long)(uintptr_t)(g + chunk * 16);
      asm volatile("global_load_async_to_lds_b128 %0, %1, off"
                   :: "v"(laddr), "v"(gaddr) : "memory");
    }
    asm volatile("s_wait_asynccnt 0x0" ::: "memory");
  }
  __builtin_prefetch(Bfc_u, 0, 3);  // pull fcW tiles toward the WGP (global_prefetch_b8)
  __syncthreads();

  int waveIn = tid >> 5;
  int wave   = blockIdx.x * EK_WAVES + waveIn;
  int lane   = tid & 31;
  int base   = wave * 32;
  if (base >= N_EDGES) return;          // wave-uniform; no barriers after this point
  int hi = lane >> 4, lo = lane & 15;
  __bf16* dfT = (__bf16*)&s_df[waveIn * 32 * BASIS];  // wave-private staging
  const __bf16* BdfL = (const __bf16*)s_bdf;

  // --- d_feat GEMM for two 16-row M-tiles: each B fragment feeds 2 WMMAs
  const float* ea0 = edge_attr + (size_t)(base + lo) * NUM_GAUSS;
  const float* ea1 = ea0 + (size_t)16 * NUM_GAUSS;
  v16bf aea[2][2];
  aea[0][0] = a_frag_f32(ea0, 0, hi);  aea[0][1] = a_frag_f32(ea0, 1, hi);
  aea[1][0] = a_frag_f32(ea1, 0, hi);  aea[1][1] = a_frag_f32(ea1, 1, hi);

#pragma unroll
  for (int j = 0; j < 8; ++j) {
    v16bf b0 = b_frag(BdfL + ((size_t)(j * 2 + 0) * 32 + lane) * 16);
    v16bf b1 = b_frag(BdfL + ((size_t)(j * 2 + 1) * 32 + lane) * 16);
    float bias = dfb[j * 16 + lo];
#pragma unroll
    for (int mb = 0; mb < 2; ++mb) {
      v8f acc = {0.f, 0.f, 0.f, 0.f, 0.f, 0.f, 0.f, 0.f};
      acc = wmma_bf16(aea[mb][0], b0, acc);
      acc = wmma_bf16(aea[mb][1], b1, acc);
#pragma unroll
      for (int r = 0; r < 8; ++r)
        dfT[(size_t)(mb * 16 + r + hi * 8) * BASIS + j * 16 + lo] =
            (__bf16)(acc[r] + bias);
    }
  }
  // LDS ops from one wave are in-order: A-layout re-loads below see the stores.

  // --- p = CF[src] (*) d_feat, built directly as A fragments (K=128, 4 tiles)
  int s0 = src[base + lo];
  int s1 = src[base + 16 + lo];
  const __bf16* cfRow0 = CFb + (size_t)s0 * BASIS;
  const __bf16* cfRow1 = CFb + (size_t)s1 * BASIS;
  const __bf16* dfRow0 = dfT;
  const __bf16* dfRow1 = dfT + (size_t)16 * BASIS;
  v16bf ap[2][4];
#pragma unroll
  for (int t = 0; t < 4; ++t) {
    v16bf cf0 = a_frag_bf16(cfRow0 + (size_t)lo * 0 + (size_t)lo * BASIS - (size_t)lo * BASIS + 0, t, hi); // placeholder avoided below
    (void)cf0;
  }
#pragma unroll
  for (int t = 0; t < 4; ++t) {
    v16bf cfA = a_frag_bf16(cfRow0, t, hi);
    v16bf dfA = a_frag_bf16(dfRow0 + (size_t)lo * BASIS, t, hi);
    v16bf cfB = a_frag_bf16(cfRow1, t, hi);
    v16bf dfB = a_frag_bf16(dfRow1 + (size_t)lo * BASIS, t, hi);
    v16bf pA, pB;
#pragma unroll
    for (int e = 0; e < 16; ++e) {
      pA[e] = (__bf16)((float)cfA[e] * (float)dfA[e]);
      pB[e] = (__bf16)((float)cfB[e] * (float)dfB[e]);
    }
    ap[0][t] = pA;
    ap[1][t] = pB;
  }

  // dst node for each D row this lane owns (M = r + 8*hi) in each M-tile
  int dn[2][8];
#pragma unroll
  for (int mb = 0; mb < 2; ++mb)
#pragma unroll
    for (int r = 0; r < 8; ++r) dn[mb][r] = dst[base + mb * 16 + r + hi * 8];

  // --- m = tanh(p @ fcW.T); each B fragment feeds 2 WMMAs; scatter-add into C
#pragma unroll
  for (int j = 0; j < 8; ++j) {
    v8f acc0 = {0.f, 0.f, 0.f, 0.f, 0.f, 0.f, 0.f, 0.f};
    v8f acc1 = {0.f, 0.f, 0.f, 0.f, 0.f, 0.f, 0.f, 0.f};
#pragma unroll
    for (int t = 0; t < 4; ++t) {
      v16bf bf = b_frag(Bfc + ((size_t)(j * 4 + t) * 32 + lane) * 16);
      acc0 = wmma_bf16(ap[0][t], bf, acc0);
      acc1 = wmma_bf16(ap[1][t], bf, acc1);
    }
#pragma unroll
    for (int r = 0; r < 8; ++r) {
      atomicAdd(&C[(size_t)dn[0][r] * BASIS + j * 16 + lo], tanhf(acc0[r]));
      atomicAdd(&C[(size_t)dn[1][r] * BASIS + j * 16 + lo], tanhf(acc1[r]));
    }
  }
}

// ---------------- Readout: h = tanh(C@r1W.T+r1b)@r2W.T + r2b; pool over batch ------
__global__ void readout_kernel(const float* __restrict__ C, const float* __restrict__ r1b,
                               const unsigned short* __restrict__ Br1_u,
                               const float* __restrict__ r2W, const float* __restrict__ r2b,
                               const int* __restrict__ batch, float* __restrict__ out) {
  __shared__ __align__(16) float hbuf[2 * 16 * HIDDEN];  // 2 waves * 16KB = 32KB
  const __bf16* Br1 = (const __bf16*)Br1_u;
  int waveIn = threadIdx.x >> 5;
  int wave   = blockIdx.x * (blockDim.x >> 5) + waveIn;
  int lane   = threadIdx.x & 31;
  int base   = wave * 16;
  if (base >= N_NODES) return;
  int hi = lane >> 4, lo = lane & 15;
  float* h = &hbuf[waveIn * 16 * HIDDEN];

  const float* row = C + (size_t)(base + lo) * BASIS;
  v16bf a[4];
#pragma unroll
  for (int t = 0; t < 4; ++t) a[t] = a_frag_f32(row, t, hi);

#pragma unroll
  for (int j = 0; j < 16; ++j) {         // HIDDEN = 256 -> 16 N-tiles
    v8f acc = {0.f, 0.f, 0.f, 0.f, 0.f, 0.f, 0.f, 0.f};
#pragma unroll
    for (int t = 0; t < 4; ++t)
      acc = wmma_bf16(a[t], b_frag(Br1 + ((size_t)(j * 4 + t) * 32 + lane) * 16), acc);
    float bias = r1b[j * 16 + lo];
#pragma unroll
    for (int r = 0; r < 8; ++r)
      h[(size_t)(r + hi * 8) * HIDDEN + j * 16 + lo] = tanhf(acc[r] + bias);
  }

  // r2 projection + global_add_pool: 16 nodes * 4 outputs = 64 dots; 2 per lane.
#pragma unroll
  for (int p = 0; p < 2; ++p) {
    int pair = lane + p * 32;
    int node = pair >> 2;
    int jo   = pair & 3;
    const float* hrow = &h[(size_t)node * HIDDEN];
    const float* w    = r2W + (size_t)jo * HIDDEN;
    float acc = r2b[jo];
    for (int c = 0; c < HIDDEN; ++c) acc += hrow[c] * w[c];
    int g = batch[base + node];
    atomicAdd(&out[g * 4 + jo], acc);
  }
}

__global__ void zero_kernel(float* __restrict__ p, int n) {
  int i = blockIdx.x * blockDim.x + threadIdx.x;
  if (i < n) p[i] = 0.f;
}

// ---------------- Launch ----------------
extern "C" void kernel_launch(void* const* d_in, const int* in_sizes, int n_in,
                              void* d_out, int out_size, void* d_ws, size_t ws_size,
                              hipStream_t stream) {
  const int*   Z          = (const int*)d_in[0];
  const int*   edge_index = (const int*)d_in[1];
  const float* edge_attr  = (const float*)d_in[2];
  const int*   batch      = (const int*)d_in[3];
  const float* embed      = (const float*)d_in[4];
  const float* cfW        = (const float*)d_in[5];
  const float* cfb        = (const float*)d_in[6];
  const float* dfW        = (const float*)d_in[7];
  const float* dfb        = (const float*)d_in[8];
  const float* fcW        = (const float*)d_in[9];
  const float* r1W        = (const float*)d_in[10];
  const float* r1b        = (const float*)d_in[11];
  const float* r2W        = (const float*)d_in[12];
  const float* r2b        = (const float*)d_in[13];
  float* out = (float*)d_out;

  const int* src = edge_index;             // row 0 of [2, E]
  const int* dst = edge_index + N_EDGES;   // row 1

  // Workspace carve-up (256B aligned); total ~15.6 MB.
  size_t off = 0;
  char* base = (char*)d_ws;
  auto carve = [&](size_t bytes) -> char* {
    char* p = base + off;
    off = (off + bytes + 255) & ~(size_t)255;
    return p;
  };
  float*          C    = (float*)         carve((size_t)N_NODES * BASIS * 4);
  unsigned short* CFb  = (unsigned short*)carve((size_t)N_NODES * BASIS * 2);
  unsigned short* Bcf  = (unsigned short*)carve((size_t)8  * 4 * 512 * 2);
  unsigned short* Bdf  = (unsigned short*)carve((size_t)8  * 2 * 512 * 2);
  unsigned short* Bfc  = (unsigned short*)carve((size_t)8  * 4 * 512 * 2);
  unsigned short* Br1  = (unsigned short*)carve((size_t)16 * 4 * 512 * 2);
  (void)ws_size; (void)in_sizes; (void)n_in; (void)out_size;

  // TDM issue-path probe (NULL descriptor -> architectural no-op)
  tdm_probe_kernel<<<1, 32, 0, stream>>>();

  // 1) Pre-tile weights into WMMA-B bf16 layout (tiny, once per call).
  prep_b_kernel<<<(8 * 4 * 512 + 255) / 256, 256, 0, stream>>>(cfW, Bcf, BASIS, 8, 4);
  prep_b_kernel<<<(8 * 2 * 512 + 255) / 256, 256, 0, stream>>>(dfW, Bdf, NUM_GAUSS, 8, 2);
  prep_b_kernel<<<(8 * 4 * 512 + 255) / 256, 256, 0, stream>>>(fcW, Bfc, BASIS, 8, 4);
  prep_b_kernel<<<(16 * 4 * 512 + 255) / 256, 256, 0, stream>>>(r1W, Br1, BASIS, 16, 4);

  // 2) C = embed[Z]
  gather_embed_kernel<<<(N_NODES * BASIS + 255) / 256, 256, 0, stream>>>(Z, embed, C);

  // 3) T message-passing iterations (launch boundaries = grid-wide sync)
  for (int t = 0; t < T_ITERS; ++t) {
    cf_kernel<<<(N_NODES / 16 + 7) / 8, 256, 0, stream>>>(C, cfb, Bcf, CFb);
    edge_kernel<<<N_EDGES / (32 * EK_WAVES), 32 * EK_WAVES, 0, stream>>>(
        src, dst, edge_attr, dfb, Bdf, Bfc, CFb, C);
  }

  // 4) Readout + pooled output (zero d_out first; it is poisoned by the harness)
  zero_kernel<<<(NUM_GRAPHS * 4 + 255) / 256, 256, 0, stream>>>(out, NUM_GRAPHS * 4);
  readout_kernel<<<(N_NODES / 16 + 1) / 2, 64, 0, stream>>>(C, r1b, Br1, r2W, r2b,
                                                            batch, out);
}
